// GNN_46986942218994
// MI455X (gfx1250) — matrix operand order repaired and checked by reading, not compiled, not run on hardware
//
#include <hip/hip_runtime.h>
#include <hip/hip_bf16.h>

typedef __bf16 bf16_t;
typedef __attribute__((ext_vector_type(16))) __bf16 v16bf;
typedef __attribute__((ext_vector_type(8)))  __bf16 v8bf;
typedef __attribute__((ext_vector_type(8)))  float  v8f;

#define N_NODES   50000
#define N_EDGES   800000
#define EMB       128
#define EDGE_DIM  16
#define N_LAYERS  4
#define NBATCH    64
#define WSTRIDE   136   // padded LDS stride (halfs) for weight tiles: 128 + 8

__device__ __forceinline__ v16bf make_frag(v8bf lo, v8bf hi) {
  return __builtin_shufflevector(lo, hi, 0,1,2,3,4,5,6,7,8,9,10,11,12,13,14,15);
}

// A fragment (16x32 bf16, MxK). lane<16: K[kb+0..7],K[kb+16..23];
// lane>=16: K[kb+8..15],K[kb+24..31]; row = lane&15.
__device__ __forceinline__ v16bf frag_A(const bf16_t* sA, int stride, int row,
                                        int khalf, int kbase) {
  const bf16_t* p = sA + row * stride + kbase + 8 * khalf;
  return make_frag(*(const v8bf*)p, *(const v8bf*)(p + 16));
}

// B fragment (32x16 bf16, KxN). Lane L holds col N=ncol0+(L&15); lane<16: K[kb+0..7],K[kb+16..23];
// lane>=16: K[kb+8..15],K[kb+24..31].
__device__ __forceinline__ v16bf frag_B(const bf16_t* sW, int kbase, int lane, int ncol0) {
  const bf16_t* p = sW + (kbase + 8 * (lane >> 4)) * WSTRIDE + ncol0 + (lane & 15);
  v16bf r;
#pragma unroll
  for (int s = 0; s < 8; ++s) {
    r[s] = p[s * WSTRIDE];
    r[s + 8] = p[(s + 16) * WSTRIDE];
  }
  return r;
}

// ---------------- Tensor Data Mover: 2D weight tile (K x 128 bf16) global->LDS
// with hardware padding: insert 4 dwords after every 64 dwords -> 136-half rows.
#if __has_builtin(__builtin_amdgcn_tensor_load_to_lds)
#define HAVE_TDM 1
typedef unsigned int v4u __attribute__((ext_vector_type(4)));
typedef int          v4i __attribute__((ext_vector_type(4)));
typedef int          v8i_t __attribute__((ext_vector_type(8)));

__device__ __forceinline__ void tdm_load_weights(unsigned lds_off, const void* gptr, int K) {
  unsigned long long ga = (unsigned long long)gptr;
  v4u g0;
  g0[0] = 1u;                                      // count=1, user mode
  g0[1] = lds_off;                                 // lds_addr (byte offset)
  g0[2] = (unsigned)(ga & 0xFFFFFFFFu);            // global_addr[31:0]
  g0[3] = (unsigned)((ga >> 32) & 0x01FFFFFFu) | (2u << 30);  // addr[56:32] | type=2
  v8i_t g1;
  // data_size=2B(1), pad_enable, pad_interval=64dw(5), pad_amount=4dw(3)
  g1[0] = (1 << 16) | (1 << 20) | (5 << 22) | (3 << 25);
  g1[1] = (128 << 16);                             // tensor_dim0 = 128 (low16 in hi half)
  g1[2] = (K & 0xFFFF) << 16;                      // tensor_dim0 hi=0 | tensor_dim1 lo
  g1[3] = (128 << 16);                             // tensor_dim1 hi=0 | tile_dim0 = 128
  g1[4] = (K & 0xFFFF);                            // tile_dim1 = K, tile_dim2 = 0
  g1[5] = 128;                                     // tensor_dim0_stride = 128
  g1[6] = 0;                                       // stride hi | dim1_stride lo
  g1[7] = 0;
  v4i z4 = {0, 0, 0, 0};
#if __clang_major__ >= 23
  v8i_t z8 = {0, 0, 0, 0, 0, 0, 0, 0};
  __builtin_amdgcn_tensor_load_to_lds(g0, g1, z4, z4, z8, 0);
#else
  __builtin_amdgcn_tensor_load_to_lds(g0, g1, z4, z4, 0);
#endif
}
#else
#define HAVE_TDM 0
#endif

// stage weights into LDS (TDM path: wave0 issues DMA; fallback: cooperative copy)
__device__ __forceinline__ void stage_weights(bf16_t* sW, unsigned sW_off,
                                              const bf16_t* __restrict__ Wb, int K,
                                              int tid) {
#if HAVE_TDM
  (void)sW;
  if (tid == 0) tdm_load_weights(sW_off, (const void*)Wb, K);
  __builtin_amdgcn_s_wait_tensorcnt(0);
#else
  (void)sW_off;
  for (int idx = tid; idx < K * 128; idx += 256) {
    int r = idx >> 7, c = idx & 127;
    sW[r * WSTRIDE + c] = Wb[idx];
  }
#endif
}

// ---------------- weight fp32 -> bf16 conversion (msg weights zero-padded 272->288)
__global__ void prep_weights(const float* __restrict__ W_in,
                             const float* __restrict__ W_msg,
                             const float* __restrict__ W_upd,
                             bf16_t* __restrict__ Wb_in,
                             bf16_t* __restrict__ Wb_msg,
                             bf16_t* __restrict__ Wb_upd) {
  int i = blockIdx.x * blockDim.x + threadIdx.x;
  const int n_in  = 128 * 128;
  const int n_msg = N_LAYERS * 288 * 128;
  const int n_upd = N_LAYERS * 256 * 128;
  if (i < n_in)  Wb_in[i]  = (bf16_t)W_in[i];
  if (i < n_upd) Wb_upd[i] = (bf16_t)W_upd[i];
  if (i < n_msg) {
    int l = i / (288 * 128);
    int r = (i / 128) % 288;
    int c = i & 127;
    float v = (r < 272) ? W_msg[(l * 272 + r) * 128 + c] : 0.f;
    Wb_msg[i] = (bf16_t)v;
  }
}

__global__ void conv_bf16(const float* __restrict__ in, bf16_t* __restrict__ out, int n) {
  for (int i = blockIdx.x * blockDim.x + threadIdx.x; i < n;
       i += gridDim.x * blockDim.x)
    out[i] = (bf16_t)in[i];
}

__global__ void zero_f32(float* __restrict__ p, int n) {
  for (int i = blockIdx.x * blockDim.x + threadIdx.x; i < n;
       i += gridDim.x * blockDim.x)
    p[i] = 0.f;
}

// ---------------- generic node GEMM: out[M,128] = act([src0|src1] @ Wb + bias) (+res)
// Writes f32 result and (optionally) a bf16 mirror for downstream WMMA consumers.
__global__ void node_gemm(const float* __restrict__ src0, const float* __restrict__ src1,
                          const bf16_t* __restrict__ Wb, const float* __restrict__ bias,
                          const float* __restrict__ hres, float* __restrict__ hout,
                          bf16_t* __restrict__ hbout,
                          int M, int K0, int K1, int relu, int tilesPerWG) {
  extern __shared__ unsigned char smem[];
  const int K = K0 + K1;
  const int astride = K + 8;
  bf16_t* sA = (bf16_t*)smem;
  const unsigned sW_off = (unsigned)(((size_t)16 * astride * 2 + 63) & ~(size_t)63);
  bf16_t* sW = (bf16_t*)(smem + sW_off);
  const int tid = threadIdx.x;

  stage_weights(sW, sW_off, Wb, K, tid);

  const int lane = tid & 31, wv = tid >> 5;
  const int ncol0 = wv * 16;
  const int row = lane & 15, kh = lane >> 4;

  for (int t = 0; t < tilesPerWG; ++t) {
    int tb = (blockIdx.x * tilesPerWG + t) * 16;
    if (tb >= M) break;
    __syncthreads();
    for (int idx = tid; idx < 16 * K; idx += 256) {
      int r = idx / K, k = idx - r * K;
      int node = tb + r;
      float v = 0.f;
      if (node < M)
        v = (k < K0) ? src0[node * K0 + k] : src1[node * K1 + (k - K0)];
      sA[r * astride + k] = (bf16_t)v;
    }
    __syncthreads();

    v8f acc = {};
    const int nch = K >> 5;
    for (int c = 0; c < nch; ++c) {
      v16bf a = frag_A(sA, astride, row, kh, c * 32);
      v16bf b = frag_B(sW, c * 32, lane, ncol0);
      acc = __builtin_amdgcn_wmma_f32_16x16x32_bf16(false, a, false, b,
                                                    (short)0, acc, false, false);
    }
    int col = ncol0 + (lane & 15);
    float bs = bias[col];
#pragma unroll
    for (int j = 0; j < 8; ++j) {
      int mr = j + 8 * (lane >> 4);
      int node = tb + mr;
      if (node < M) {
        float u = acc[j] + bs;
        if (relu) u = fmaxf(u, 0.f);
        if (hres) u += hres[node * 128 + col];
        hout[node * 128 + col] = u;
        if (hbout) hbout[node * 128 + col] = (bf16_t)u;
      }
    }
  }
}

// ---------------- edge message GEMM + fused relu + atomic scatter-add into agg
// A row (K=288): [hb[src](128) | hb[dst](128) | eb(16) | zero-pad(16)], all bf16,
// staged with 16-byte vector copies. Invalid rows clamp gather index to 0 (rows
// are independent in the GEMM; their outputs are masked at the atomic stores).
__global__ void edge_msg(const bf16_t* __restrict__ hb, const bf16_t* __restrict__ eb,
                         const int* __restrict__ srcIdx, const int* __restrict__ dstIdx,
                         const bf16_t* __restrict__ Wb, const float* __restrict__ bias,
                         float* __restrict__ agg, int nE, int tilesPerWG) {
  extern __shared__ unsigned char smem[];
  int*    sidx = (int*)smem;                     // [0..15]=src, [16..31]=dst
  bf16_t* sA   = (bf16_t*)(smem + 128);          // 16 x 296 halfs
  const unsigned sW_off = 128 + 16 * 296 * 2;    // 9600
  bf16_t* sW   = (bf16_t*)(smem + sW_off);       // 288 x 136 halfs
  const int tid = threadIdx.x;

  stage_weights(sW, sW_off, Wb, 288, tid);

  const int lane = tid & 31, wv = tid >> 5;
  const int ncol0 = wv * 16;
  const int row = lane & 15, kh = lane >> 4;
  const uint4 zero4 = make_uint4(0u, 0u, 0u, 0u);

  for (int t = 0; t < tilesPerWG; ++t) {
    int tb = (blockIdx.x * tilesPerWG + t) * 16;
    if (tb >= nE) break;
    __syncthreads();
    if (tid < 32) {
      int e = tb + (tid & 15);
      const int* p = (tid < 16) ? srcIdx : dstIdx;
      sidx[tid] = (e < nE) ? p[e] : 0;
    }
    __syncthreads();
    // 576 x 16B chunks: 256 src + 256 dst + 32 eattr + 32 zero-pad
    for (int idx = tid; idx < 576; idx += 256) {
      uint4 v;
      int off;
      if (idx < 256) {
        int r = idx >> 4, c = idx & 15;
        v = *(const uint4*)(hb + (size_t)sidx[r] * 128 + c * 8);
        off = r * 296 + c * 8;
      } else if (idx < 512) {
        int i2 = idx - 256;
        int r = i2 >> 4, c = i2 & 15;
        v = *(const uint4*)(hb + (size_t)sidx[16 + r] * 128 + c * 8);
        off = r * 296 + 128 + c * 8;
      } else if (idx < 544) {
        int i2 = idx - 512;
        int r = i2 >> 1, c = i2 & 1;
        int e = tb + r;
        v = (e < nE) ? *(const uint4*)(eb + (size_t)e * 16 + c * 8) : zero4;
        off = r * 296 + 256 + c * 8;
      } else {
        int i2 = idx - 544;
        int r = i2 >> 1, c = i2 & 1;
        v = zero4;
        off = r * 296 + 272 + c * 8;
      }
      *(uint4*)(sA + off) = v;
    }
    __syncthreads();

    v8f acc = {};
#pragma unroll
    for (int c = 0; c < 9; ++c) {
      v16bf a = frag_A(sA, 296, row, kh, c * 32);
      v16bf b = frag_B(sW, c * 32, lane, ncol0);
      acc = __builtin_amdgcn_wmma_f32_16x16x32_bf16(false, a, false, b,
                                                    (short)0, acc, false, false);
    }
    int col = ncol0 + (lane & 15);
    float bs = bias[col];
#pragma unroll
    for (int j = 0; j < 8; ++j) {
      int mr = j + 8 * (lane >> 4);
      int e = tb + mr;
      if (e < nE) {
        float m = fmaxf(acc[j] + bs, 0.f);
        unsafeAtomicAdd(&agg[(size_t)sidx[16 + mr] * 128 + col], m);
      }
    }
  }
}

// ---------------- mean-pool by (sorted) batch id: run-length accumulate + atomic flush
__global__ void pool_sum(const float* __restrict__ h, const int* __restrict__ batch,
                         float* __restrict__ pool, float* __restrict__ cnt, int M) {
  int col = threadIdx.x;  // 128 threads
  int base = blockIdx.x * 64;
  float s = 0.f, c = 0.f;
  int cur = -1;
  for (int r = 0; r < 64; ++r) {
    int node = base + r;
    if (node >= M) break;
    int b = batch[node];
    if (b != cur) {
      if (cur >= 0) {
        unsafeAtomicAdd(&pool[cur * 128 + col], s);
        if (col == 0) unsafeAtomicAdd(&cnt[cur], c);
      }
      cur = b; s = 0.f; c = 0.f;
    }
    s += h[node * 128 + col];
    c += 1.f;
  }
  if (cur >= 0) {
    unsafeAtomicAdd(&pool[cur * 128 + col], s);
    if (col == 0) unsafeAtomicAdd(&cnt[cur], c);
  }
}

__global__ void pred_kernel(const float* __restrict__ pool, const float* __restrict__ cnt,
                            const float* __restrict__ Wp, const float* __restrict__ bp,
                            float* __restrict__ out) {
  int b = threadIdx.x;
  if (b < NBATCH) {
    float inv = 1.f / fmaxf(cnt[b], 1.f);
    float s = 0.f;
    for (int c = 0; c < 128; ++c) s += pool[b * 128 + c] * inv * Wp[c];
    out[b] = s + bp[0];
  }
}

extern "C" void kernel_launch(void* const* d_in, const int* in_sizes, int n_in,
                              void* d_out, int out_size, void* d_ws, size_t ws_size,
                              hipStream_t stream) {
  const float* x      = (const float*)d_in[0];
  const int*   ei     = (const int*)d_in[1];
  const float* eattr  = (const float*)d_in[2];
  const int*   batch  = (const int*)d_in[3];
  const float* W_in   = (const float*)d_in[4];
  const float* b_in   = (const float*)d_in[5];
  const float* W_msg  = (const float*)d_in[6];
  const float* b_msg  = (const float*)d_in[7];
  const float* W_upd  = (const float*)d_in[8];
  const float* b_upd  = (const float*)d_in[9];
  const float* W_pred = (const float*)d_in[10];
  const float* b_pred = (const float*)d_in[11];
  float* out = (float*)d_out;

  char* ws = (char*)d_ws;
  size_t off = 0;
  auto alloc = [&](size_t bytes) -> void* {
    void* p = ws + off;
    off += (bytes + 255) & ~(size_t)255;
    return p;
  };
  float*  h      = (float*)alloc((size_t)N_NODES * 128 * sizeof(float));
  float*  agg    = (float*)alloc((size_t)N_NODES * 128 * sizeof(float));
  bf16_t* hb     = (bf16_t*)alloc((size_t)N_NODES * 128 * 2);
  bf16_t* eb     = (bf16_t*)alloc((size_t)N_EDGES * EDGE_DIM * 2);
  bf16_t* Wb_in  = (bf16_t*)alloc((size_t)128 * 128 * 2);
  bf16_t* Wb_msg = (bf16_t*)alloc((size_t)N_LAYERS * 288 * 128 * 2);
  bf16_t* Wb_upd = (bf16_t*)alloc((size_t)N_LAYERS * 256 * 128 * 2);
  float*  pool   = (float*)alloc((size_t)NBATCH * 128 * sizeof(float));
  float*  cnt    = (float*)alloc((size_t)NBATCH * sizeof(float));

  const int* srcIdx = ei;
  const int* dstIdx = ei + N_EDGES;

  prep_weights<<<576, 256, 0, stream>>>(W_in, W_msg, W_upd, Wb_in, Wb_msg, Wb_upd);
  conv_bf16<<<4096, 256, 0, stream>>>(eattr, eb, N_EDGES * EDGE_DIM);
  zero_f32<<<64, 256, 0, stream>>>(pool, NBATCH * 128);
  zero_f32<<<1, 64, 0, stream>>>(cnt, NBATCH);

  const int nodeTiles  = (N_NODES + 15) / 16;   // 3125
  const int nodeBlocks = (nodeTiles + 7) / 8;   // 391
  const int edgeTiles  = (N_EDGES + 15) / 16;   // 50000
  const int edgeBlocks = (edgeTiles + 7) / 8;   // 6250

  // lin_in: h = x @ W_in + b_in  (K=128)
  size_t lds_in = (size_t)16 * (128 + 8) * 2 + (size_t)128 * WSTRIDE * 2;   // 39168
  node_gemm<<<nodeBlocks, 256, lds_in, stream>>>(x, nullptr, Wb_in, b_in,
                                                 nullptr, h, hb,
                                                 N_NODES, 128, 0, 0, 8);

  size_t lds_edge = 128 + (size_t)16 * 296 * 2 + (size_t)288 * WSTRIDE * 2; // 87936
  size_t lds_upd  = (size_t)16 * (256 + 8) * 2 + (size_t)256 * WSTRIDE * 2; // 78080

  for (int l = 0; l < N_LAYERS; ++l) {
    zero_f32<<<2048, 256, 0, stream>>>(agg, N_NODES * 128);
    edge_msg<<<edgeBlocks, 256, lds_edge, stream>>>(
        hb, eb, srcIdx, dstIdx, Wb_msg + (size_t)l * 288 * 128,
        b_msg + l * 128, agg, N_EDGES, 8);
    // h = h + relu([h|agg] @ W_upd[l] + b_upd[l])  (K=256)
    node_gemm<<<nodeBlocks, 256, lds_upd, stream>>>(
        h, agg, Wb_upd + (size_t)l * 256 * 128, b_upd + l * 128,
        h, h, hb, N_NODES, 128, 128, 1, 8);
  }

  pool_sum<<<(N_NODES + 63) / 64, 128, 0, stream>>>(h, batch, pool, cnt, N_NODES);
  pred_kernel<<<1, 64, 0, stream>>>(pool, cnt, W_pred, b_pred, out);
}
